// Attention_64175401337442
// MI455X (gfx1250) — compile-verified, matrix operand order checked by reading, not run on hardware
//
#include <hip/hip_runtime.h>

typedef __bf16 bf16;
typedef __bf16 v16bf __attribute__((ext_vector_type(16)));
typedef __bf16 v8bf  __attribute__((ext_vector_type(8)));
typedef float  v8f   __attribute__((ext_vector_type(8)));

typedef unsigned int u32x4 __attribute__((ext_vector_type(4)));
typedef int          i32x4 __attribute__((ext_vector_type(4)));
typedef int          i32x8 __attribute__((ext_vector_type(8)));

#define N_TOK   65536
#define C_DIM   512
#define H_DIM   8
#define K_TOK   256
#define D_HEAD  64
#define QKV_COLS 1536
#define ATT_SCALE 0.125f

#if defined(__has_builtin)
#  if __has_builtin(__builtin_amdgcn_tensor_load_to_lds) && \
      __has_builtin(__builtin_amdgcn_s_wait_tensorcnt)
#    define HAVE_TDM 1
#  endif
#endif
#ifndef HAVE_TDM
#  define HAVE_TDM 0
#endif

// ---------- helpers ----------

__device__ __forceinline__ v8f wmma_bf16(v16bf a, v16bf b, v8f c) {
  return __builtin_amdgcn_wmma_f32_16x16x32_bf16(false, a, false, b, (short)0, c,
                                                 false, false);
}

// 16-bit A/B tile lane layout for 16x16x32:
//   lanes 0-15 : K = {0..7, 16..23}; lanes 16-31 : K = {8..15, 24..31}
// 'row' points at element (matrix_row, 0); rows must be 16B-aligned.
__device__ __forceinline__ v16bf tile16(const bf16* row, int half) {
  const bf16* p = row + (half << 3);
  v8bf lo = *reinterpret_cast<const v8bf*>(p);
  v8bf hi = *reinterpret_cast<const v8bf*>(p + 16);
  v16bf r;
#pragma unroll
  for (int i = 0; i < 8; ++i) { r[i] = lo[i]; r[i + 8] = hi[i]; }
  return r;
}

// Same layout, but source is fp32 (converted to bf16 in registers).
__device__ __forceinline__ v16bf tile16_f32(const float* row, int half) {
  const float* p = row + (half << 3);
  v16bf r;
#pragma unroll
  for (int i = 0; i < 8; ++i) {
    r[i]     = (bf16)p[i];
    r[i + 8] = (bf16)p[16 + i];
  }
  return r;
}

// butterfly reduction inside a 16-lane half-wave (bit4 of lane untouched)
__device__ __forceinline__ float half_max(float v) {
#pragma unroll
  for (int m = 1; m <= 8; m <<= 1) v = fmaxf(v, __shfl_xor(v, m, 32));
  return v;
}
__device__ __forceinline__ float half_sum(float v) {
#pragma unroll
  for (int m = 1; m <= 8; m <<= 1) v += __shfl_xor(v, m, 32);
  return v;
}

// ---------- kernel 1: qkv = feat[order] @ Wqkv + bqkv   (bf16 out) ----------

#define BT_STRIDE 48   // 96B rows: 16B-aligned chunks

__global__ __launch_bounds__(128)
void qkv_gemm(const float* __restrict__ feat, const long long* __restrict__ order,
              const float* __restrict__ Wqkv, const float* __restrict__ bqkv,
              bf16* __restrict__ qkvp) {
  __shared__ bf16 Bt[64 * BT_STRIDE];

  const int n0   = blockIdx.x * 64;
  const int m0   = blockIdx.y * 64;
  const int tid  = threadIdx.x;
  const int lane = tid & 31;
  const int wave = tid >> 5;
  const int half = (lane >> 4) & 1;
  const int l16  = lane & 15;

  const long long grow = order[m0 + wave * 16 + l16];
  const float* arow = feat + (size_t)grow * C_DIM;

  v8f acc[4];
#pragma unroll
  for (int j = 0; j < 4; ++j) acc[j] = (v8f)0.0f;

  for (int k0 = 0; k0 < C_DIM; k0 += 32) {
    // stage B tile (32 x 64 of Wqkv) transposed as bf16: Bt[n][k]
#pragma unroll
    for (int i = 0; i < 4; ++i) {
      int idx = i * 128 + tid;            // 512 float4 slots
      int kk = idx >> 4;                  // 0..31
      int nn = (idx & 15) * 4;            // 0..60
      const float4 w =
          *reinterpret_cast<const float4*>(&Wqkv[(size_t)(k0 + kk) * QKV_COLS + n0 + nn]);
      Bt[(nn + 0) * BT_STRIDE + kk] = (bf16)w.x;
      Bt[(nn + 1) * BT_STRIDE + kk] = (bf16)w.y;
      Bt[(nn + 2) * BT_STRIDE + kk] = (bf16)w.z;
      Bt[(nn + 3) * BT_STRIDE + kk] = (bf16)w.w;
    }
    __syncthreads();

    v16bf a = tile16_f32(arow + k0, half);
#pragma unroll
    for (int j = 0; j < 4; ++j) {
      v16bf b = tile16(Bt + (16 * j + l16) * BT_STRIDE, half);
      acc[j] = wmma_bf16(a, b, acc[j]);
    }
    __syncthreads();
  }

  // epilogue: C layout -> row = r + 8*half, col = 16*j + l16
#pragma unroll
  for (int j = 0; j < 4; ++j) {
    int col = n0 + 16 * j + l16;
    float bias = bqkv[col];
#pragma unroll
    for (int r = 0; r < 8; ++r) {
      int row = m0 + wave * 16 + r + 8 * half;
      qkvp[(size_t)row * QKV_COLS + col] = (bf16)(acc[j][r] + bias);
    }
  }
}

// ---------- kernel 2: windowed flash attention per (p,h) ----------

#define KS_STRIDE 80    // 160B rows (= 128B data + 32B TDM pad)
#define VT_STRIDE 272   // 544B rows
#define PR_STRIDE 48    // 96B rows
#define SM_ELEMS (K_TOK * KS_STRIDE + D_HEAD * VT_STRIDE + 16 * 16 * PR_STRIDE)

__global__ __launch_bounds__(512)
void attention(const bf16* __restrict__ qkvp, bf16* __restrict__ attnout) {
  extern __shared__ bf16 smem[];
  bf16* Ks = smem;                                    // [256][KS_STRIDE]
  bf16* Vt = Ks + K_TOK * KS_STRIDE;                  // [64][VT_STRIDE] (transposed V)
  bf16* Pr = Vt + D_HEAD * VT_STRIDE;                 // per-wave [16][PR_STRIDE]

  const int p = blockIdx.x >> 3;
  const int h = blockIdx.x & 7;
  const size_t gp = (size_t)p * K_TOK;

  const int tid  = threadIdx.x;
  const int lane = tid & 31;
  const int wave = tid >> 5;          // 0..15
  const int half = (lane >> 4) & 1;
  const int l16  = lane & 15;

  const bf16* kbase = qkvp + gp * QKV_COLS + C_DIM + h * D_HEAD;
  const bf16* vbase = qkvp + gp * QKV_COLS + 2 * C_DIM + h * D_HEAD;

#if HAVE_TDM
  // Tensor Data Mover: 2D tile 64 x 256 (bf16), global row stride 1536 elem,
  // LDS padded to 80-element rows via pad_interval=32 DW / pad_amount=8 DW.
  if (wave == 0) {
    unsigned long long ga = (unsigned long long)(uintptr_t)kbase;
    unsigned int ldsa = (unsigned int)(uintptr_t)Ks;   // low 32 bits = LDS offset
    u32x4 g0;
    g0[0] = 1u;                                        // count=1, user descriptor
    g0[1] = ldsa;                                      // lds_addr
    g0[2] = (unsigned int)ga;                          // global_addr[31:0]
    g0[3] = (unsigned int)((ga >> 32) & 0x1FFFFFFu) | (2u << 30); // addr[56:32]|type=2
    i32x8 g1;
    g1[0] = (int)((1u << 16)   // data_size = 2B
                | (1u << 20)   // pad_enable
                | (4u << 22)   // pad_interval: 32 DWORDs (one 64-elem row)
                | (7u << 25)); // pad_amount: 8 DWORDs (16 elems)
    g1[1] = (int)(64u << 16);              // tensor_dim0 = 64
    g1[2] = (int)(256u << 16);             // tensor_dim1 = 256
    g1[3] = (int)(64u << 16);              // tile_dim0 = 64
    g1[4] = 256;                           // tile_dim1 = 256, tile_dim2 = 0
    g1[5] = QKV_COLS;                      // tensor_dim0_stride = 1536
    g1[6] = 0;
    g1[7] = 0;
    i32x4 gz = {0, 0, 0, 0};
#if __clang_major__ >= 23
    i32x8 gz8 = {0, 0, 0, 0, 0, 0, 0, 0};
    __builtin_amdgcn_tensor_load_to_lds(g0, g1, gz, gz, gz8, 0);
#else
    __builtin_amdgcn_tensor_load_to_lds(g0, g1, gz, gz, 0);
#endif
    __builtin_amdgcn_s_wait_tensorcnt((short)0);
  }
#endif

  // stage V transposed (and K on the non-TDM path) with b128 global loads
#pragma unroll
  for (int i = 0; i < 4; ++i) {
    int idx = i * 512 + tid;               // 2048 v8bf slots
    int kr = idx >> 3;
    int dd = (idx & 7) * 8;
    v8bf vv = *reinterpret_cast<const v8bf*>(vbase + (size_t)kr * QKV_COLS + dd);
#pragma unroll
    for (int j = 0; j < 8; ++j) Vt[(dd + j) * VT_STRIDE + kr] = vv[j];
#if !HAVE_TDM
    v8bf kk8 = *reinterpret_cast<const v8bf*>(kbase + (size_t)kr * QKV_COLS + dd);
    *reinterpret_cast<v8bf*>(Ks + kr * KS_STRIDE + dd) = kk8;
#endif
  }
  __syncthreads();

  // this wave owns query rows [16*wave, 16*wave+16)
  const int rowbase = wave * 16;
  const bf16* qrow = qkvp + (gp + rowbase + l16) * QKV_COLS + h * D_HEAD;
  v16bf qa0 = tile16(qrow, half);        // Q k-dims 0..31
  v16bf qa1 = tile16(qrow + 32, half);   // Q k-dims 32..63

  float rmax[8], rsum[8];
#pragma unroll
  for (int r = 0; r < 8; ++r) { rmax[r] = -3.0e38f; rsum[r] = 0.0f; }
  v8f oacc[4];
#pragma unroll
  for (int j = 0; j < 4; ++j) oacc[j] = (v8f)0.0f;

  bf16* myPr = Pr + wave * 16 * PR_STRIDE;

  for (int kc = 0; kc < 8; ++kc) {        // 32 keys per chunk
    v8f s[2];
#pragma unroll
    for (int t = 0; t < 2; ++t) {
      s[t] = (v8f)0.0f;
      const bf16* krow = Ks + (kc * 32 + t * 16 + l16) * KS_STRIDE;
      v16bf b0 = tile16(krow, half);       // K^T k-dims 0..31
      v16bf b1 = tile16(krow + 32, half);  // K^T k-dims 32..63
      s[t] = wmma_bf16(qa0, b0, s[t]);
      s[t] = wmma_bf16(qa1, b1, s[t]);
    }

    // online softmax update (rows of C layout == VGPR index, per half-wave)
    float f[8];
#pragma unroll
    for (int r = 0; r < 8; ++r) {
      float cm = half_max(fmaxf(s[0][r], s[1][r]));
      float nm = fmaxf(rmax[r], cm);
      f[r] = __expf((rmax[r] - nm) * ATT_SCALE);
      rmax[r] = nm;
    }

    float ps[8];
#pragma unroll
    for (int r = 0; r < 8; ++r) ps[r] = 0.0f;
#pragma unroll
    for (int t = 0; t < 2; ++t) {
#pragma unroll
      for (int r = 0; r < 8; ++r) {
        float pv = __expf((s[t][r] - rmax[r]) * ATT_SCALE);
        ps[r] += pv;
        myPr[(r + 8 * half) * PR_STRIDE + t * 16 + l16] = (bf16)pv;
      }
    }
#pragma unroll
    for (int r = 0; r < 8; ++r) {
      rsum[r] = rsum[r] * f[r] + half_sum(ps[r]);
#pragma unroll
      for (int j = 0; j < 4; ++j) oacc[j][r] *= f[r];
    }

    // probs back through LDS as a 16x32 bf16 A tile (same-wave DS ordering)
    v16bf pa = tile16(myPr + l16 * PR_STRIDE, half);
#pragma unroll
    for (int j = 0; j < 4; ++j) {
      v16bf vb = tile16(Vt + (16 * j + l16) * VT_STRIDE + kc * 32, half);
      oacc[j] = wmma_bf16(pa, vb, oacc[j]);
    }
  }

  // epilogue: divide by row sums, store bf16 (permuted token space)
  float rinv[8];
#pragma unroll
  for (int r = 0; r < 8; ++r) rinv[r] = 1.0f / rsum[r];
#pragma unroll
  for (int j = 0; j < 4; ++j) {
    int col = h * D_HEAD + 16 * j + l16;
#pragma unroll
    for (int r = 0; r < 8; ++r) {
      size_t row = gp + rowbase + r + 8 * half;
      attnout[row * C_DIM + col] = (bf16)(oacc[j][r] * rinv[r]);
    }
  }
}

// ---------- kernel 3: out = attnout[inverse] @ Wproj + bproj  (f32 out) ----------

__global__ __launch_bounds__(128)
void proj_gemm(const bf16* __restrict__ attnout, const long long* __restrict__ inverse,
               const float* __restrict__ Wproj, const float* __restrict__ bproj,
               float* __restrict__ out) {
  __shared__ bf16 Bt[64 * BT_STRIDE];

  const int n0   = blockIdx.x * 64;
  const int m0   = blockIdx.y * 64;
  const int tid  = threadIdx.x;
  const int lane = tid & 31;
  const int wave = tid >> 5;
  const int half = (lane >> 4) & 1;
  const int l16  = lane & 15;

  const long long grow = inverse[m0 + wave * 16 + l16];
  const bf16* arow = attnout + (size_t)grow * C_DIM;

  v8f acc[4];
#pragma unroll
  for (int j = 0; j < 4; ++j) acc[j] = (v8f)0.0f;

  for (int k0 = 0; k0 < C_DIM; k0 += 32) {
#pragma unroll
    for (int i = 0; i < 4; ++i) {
      int idx = i * 128 + tid;
      int kk = idx >> 4;
      int nn = (idx & 15) * 4;
      const float4 w =
          *reinterpret_cast<const float4*>(&Wproj[(size_t)(k0 + kk) * C_DIM + n0 + nn]);
      Bt[(nn + 0) * BT_STRIDE + kk] = (bf16)w.x;
      Bt[(nn + 1) * BT_STRIDE + kk] = (bf16)w.y;
      Bt[(nn + 2) * BT_STRIDE + kk] = (bf16)w.z;
      Bt[(nn + 3) * BT_STRIDE + kk] = (bf16)w.w;
    }
    __syncthreads();

    v16bf a = tile16(arow + k0, half);
#pragma unroll
    for (int j = 0; j < 4; ++j) {
      v16bf b = tile16(Bt + (16 * j + l16) * BT_STRIDE, half);
      acc[j] = wmma_bf16(a, b, acc[j]);
    }
    __syncthreads();
  }

#pragma unroll
  for (int j = 0; j < 4; ++j) {
    int col = n0 + 16 * j + l16;
    float bias = bproj[col];
#pragma unroll
    for (int r = 0; r < 8; ++r) {
      int row = m0 + wave * 16 + r + 8 * half;
      out[(size_t)row * C_DIM + col] = acc[j][r] + bias;
    }
  }
}

// ---------- launch ----------

extern "C" void kernel_launch(void* const* d_in, const int* in_sizes, int n_in,
                              void* d_out, int out_size, void* d_ws, size_t ws_size,
                              hipStream_t stream) {
  const float*     feat    = (const float*)d_in[0];
  const long long* order   = (const long long*)d_in[1];
  const long long* inverse = (const long long*)d_in[2];
  const float*     Wqkv    = (const float*)d_in[3];
  const float*     bqkv    = (const float*)d_in[4];
  const float*     Wproj   = (const float*)d_in[5];
  const float*     bproj   = (const float*)d_in[6];
  float* out = (float*)d_out;

  bf16* qkvp    = (bf16*)d_ws;                                   // 192 MB
  bf16* attnout = (bf16*)((char*)d_ws + (size_t)N_TOK * QKV_COLS * sizeof(bf16)); // 64 MB

  qkv_gemm<<<dim3(QKV_COLS / 64, N_TOK / 64), 128, 0, stream>>>(
      feat, order, Wqkv, bqkv, qkvp);

  size_t smem = (size_t)SM_ELEMS * sizeof(bf16);   // ~98 KB (<=320 KB/WGP)
  attention<<<dim3((N_TOK / K_TOK) * H_DIM), 512, smem, stream>>>(qkvp, attnout);

  proj_gemm<<<dim3(C_DIM / 64, N_TOK / 64), 128, 0, stream>>>(
      attnout, inverse, Wproj, bproj, out);
}
